// TinyCodec_67894843015556
// MI455X (gfx1250) — matrix operand order repaired and checked by reading, not compiled, not run on hardware
//
#include <hip/hip_runtime.h>
#include <hip/hip_bf16.h>

typedef _Float16 half_t;
typedef __attribute__((ext_vector_type(16))) _Float16 v16h;
typedef __attribute__((ext_vector_type(8)))  float    v8f;
typedef __attribute__((ext_vector_type(4)))  unsigned int u32x4;
typedef __attribute__((ext_vector_type(8)))  unsigned int u32x8;

#define B_      32
#define T_IN    16384
#define T2      8192      // after conv1 (stride2)
#define T4      4096      // after conv2 (stride2)
#define ZDIM    64
#define HID     32
#define KCODES  1024
#define NROWS   (B_ * T4)   // 131072

__device__ __forceinline__ float lrelu(float v) { return v > 0.f ? v : 0.2f * v; }

// ---------------------------------------------------------------- prep: cbT (f16, K-major) + codeNorm
__global__ __launch_bounds__(256)
void prep_codebook(const float* __restrict__ cb, half_t* __restrict__ cbT,
                   float* __restrict__ codeNorm) {
    int n = blockIdx.x * 256 + threadIdx.x;      // 1024 codes
    if (n >= KCODES) return;
    float norm = 0.f;
    #pragma unroll 8
    for (int d = 0; d < ZDIM; ++d) {
        float v = cb[n * ZDIM + d];
        norm += v * v;
        cbT[d * KCODES + n] = (half_t)v;
    }
    codeNorm[n] = norm;
}

// ---------------------------------------------------------------- conv1: 1->32, K15, s2, p7 + lrelu
__global__ __launch_bounds__(256)
void conv1_kernel(const float* __restrict__ x, const float* __restrict__ w,
                  const float* __restrict__ bias, float* __restrict__ out) {
    __shared__ float lw[HID * 15];
    __shared__ float lb[HID];
    int tid = threadIdx.x;
    for (int i = tid; i < HID * 15; i += 256) lw[i] = w[i];
    if (tid < HID) lb[tid] = bias[tid];
    __syncthreads();
    int gid = blockIdx.x * 256 + tid;            // (b*32 + o)*8192 + t
    int t = gid & (T2 - 1);
    int bo = gid >> 13;
    int o = bo & (HID - 1);
    int b = bo >> 5;
    const float* xb = x + b * T_IN;
    float acc = lb[o];
    int base = 2 * t - 7;
    #pragma unroll
    for (int k = 0; k < 15; ++k) {
        int p = base + k;
        float xv = (p >= 0 && p < T_IN) ? xb[p] : 0.f;
        acc = fmaf(lw[o * 15 + k], xv, acc);
    }
    out[gid] = lrelu(acc);
}

// ---------------------------------------------------------------- conv2: 32->64, K15, s2, p7 + lrelu
// grid: (t-tiles=256, o-groups=4, b=32); block 256 = 16 o x 16 t
__global__ __launch_bounds__(256)
void conv2_kernel(const float* __restrict__ h1, const float* __restrict__ w,
                  const float* __restrict__ bias, float* __restrict__ out) {
    __shared__ float lw[16 * HID * 15];          // 30 KB
    __shared__ float xt[HID][48];                // positions 2*t0-7 .. 2*t0+37 (45 used)
    int tid = threadIdx.x;
    int t0 = blockIdx.x * 16;
    int og = blockIdx.y;
    int b  = blockIdx.z;
    for (int id = tid; id < 16 * HID * 15; id += 256) lw[id] = w[og * (16 * HID * 15) + id];
    for (int id = tid; id < HID * 45; id += 256) {
        int ch = id / 45, j = id % 45;
        int p = 2 * t0 - 7 + j;
        xt[ch][j] = (p >= 0 && p < T2) ? h1[(b * HID + ch) * T2 + p] : 0.f;
    }
    __syncthreads();
    int oy = tid >> 4, tx = tid & 15;
    int o = og * 16 + oy;
    float acc = bias[o];
    #pragma unroll 4
    for (int i = 0; i < HID; ++i) {
        const float* wp = &lw[(oy * HID + i) * 15];
        #pragma unroll
        for (int k = 0; k < 15; ++k)
            acc = fmaf(wp[k], xt[i][2 * tx + k], acc);
    }
    out[(b * (2 * HID) + o) * T4 + t0 + tx] = lrelu(acc);
}

// ---------------------------------------------------------------- conv3: 64->64, K3, s1, p1 -> zh (f16 row-major) + rowNorm
// grid: (t-tiles=32, b=32); block 128 threads (one t each)
__global__ __launch_bounds__(128)
void conv3_kernel(const float* __restrict__ h2, const float* __restrict__ w,
                  const float* __restrict__ bias, half_t* __restrict__ zh,
                  float* __restrict__ rowNorm) {
    __shared__ float xt[ZDIM][130];              // t0-1 .. t0+128
    int tid = threadIdx.x;
    int t0 = blockIdx.x * 128;
    int b  = blockIdx.y;
    for (int id = tid; id < ZDIM * 130; id += 128) {
        int ch = id / 130, j = id % 130;
        int p = t0 - 1 + j;
        xt[ch][j] = (p >= 0 && p < T4) ? h2[(b * ZDIM + ch) * T4 + p] : 0.f;
    }
    __syncthreads();
    int t = t0 + tid;
    int row = b * T4 + t;
    float norm = 0.f;
    for (int d = 0; d < ZDIM; ++d) {
        float acc = bias[d];                      // uniform -> scalar loads
        #pragma unroll 8
        for (int i = 0; i < ZDIM; ++i) {
            const float* wp = w + (d * ZDIM + i) * 3;   // uniform indices
            acc = fmaf(wp[0], xt[i][tid], acc);
            acc = fmaf(wp[1], xt[i][tid + 1], acc);
            acc = fmaf(wp[2], xt[i][tid + 2], acc);
        }
        zh[row * ZDIM + d] = (half_t)acc;
        norm = fmaf(acc, acc, norm);
    }
    rowNorm[row] = norm;
}

// ---------------------------------------------------------------- VQ: WMMA distance argmin
// block 256 = 8 waves, each wave owns 16 rows; LDS-resident f16 codebook (K-major)
// staged by the Tensor Data Mover (one tensor_load_to_lds per block)
__global__ __launch_bounds__(256)
void vq_kernel(const half_t* __restrict__ zh, const float* __restrict__ rowNorm,
               const half_t* __restrict__ cbT, const float* __restrict__ codeNorm,
               int* __restrict__ idxOut, float* __restrict__ idxF,
               float* __restrict__ lossSum) {
    __shared__ half_t lcb[ZDIM * KCODES];        // 128 KB (CDNA5: 320 KB/WGP)
    __shared__ float  lcn[KCODES];               // 4 KB
    int tid = threadIdx.x;

    if (tid < 32) {
        // --- TDM: DMA the whole 64x1024 f16 tile (128 KB) global -> LDS ---
        unsigned ldsOff = (unsigned)(size_t)(void*)lcb;
        unsigned long long ga = (unsigned long long)(size_t)cbT;
        u32x4 g0;
        g0[0] = 1u;                                             // count=1, user D#
        g0[1] = ldsOff;                                         // lds_addr
        g0[2] = (unsigned)(ga & 0xFFFFFFFFu);                   // global_addr[31:0]
        g0[3] = (unsigned)((ga >> 32) & 0x01FFFFFFu) | (2u << 30); // addr[56:32] | type=2
        u32x8 g1;
        g1[0] = 0x00010000u;                    // wg_mask=0, data_size=2B
        g1[1] = (KCODES & 0xFFFFu) << 16;       // tensor_dim0 = 1024 (bits 63:48)
        g1[2] = (ZDIM & 0xFFFFu) << 16;         // tensor_dim1 = 64   (bits 95:80)
        g1[3] = (KCODES & 0xFFFFu) << 16;       // tile_dim0 = 1024   (bits 127:112)
        g1[4] = (unsigned)ZDIM;                 // tile_dim1 = 64 (tile_dim2 = 0)
        g1[5] = (unsigned)KCODES;               // tensor_dim0_stride = 1024
        g1[6] = 0u;
        g1[7] = 0u;
        asm volatile("tensor_load_to_lds %0, %1" :: "s"(g0), "s"(g1) : "memory");
        __builtin_amdgcn_s_wait_tensorcnt(0);
    }
    for (int i = tid; i < KCODES; i += 256) lcn[i] = codeNorm[i];
    __syncthreads();

    int wave = tid >> 5;
    int lane = tid & 31;
    int halfSel = lane >> 4;                     // 0: lanes 0-15, 1: lanes 16-31
    int laneN = lane & 15;
    int m0 = blockIdx.x * 128 + wave * 16;

    // ---- A operand: 16 rows x 64 (f16) split into two K=32 chunks (ISA layout)
    const half_t* zrow = zh + (size_t)(m0 + laneN) * ZDIM;
    int kb = halfSel ? 8 : 0;                    // lanes>=16 hold K 8-15 / 24-31
    v16h a0, a1;
    *((uint4*)&a0)     = *(const uint4*)(zrow + kb);
    *((uint4*)&a0 + 1) = *(const uint4*)(zrow + kb + 16);
    *((uint4*)&a1)     = *(const uint4*)(zrow + 32 + kb);
    *((uint4*)&a1 + 1) = *(const uint4*)(zrow + 32 + kb + 16);

    float rn[8];
    #pragma unroll
    for (int r = 0; r < 8; ++r) rn[r] = rowNorm[m0 + r + 8 * halfSel];

    float minD[8];
    int   minI[8];
    #pragma unroll
    for (int r = 0; r < 8; ++r) { minD[r] = 3.0e38f; minI[r] = 0; }

    for (int tile = 0; tile < KCODES / 16; ++tile) {
        // B operand: lane = N, elem = K (same K order as A, strided f16 reads from LDS)
        const half_t* bp = lcb + tile * 16 + laneN;
        v16h b0, b1;
        #pragma unroll
        for (int s = 0; s < 16; ++s) {
            int k = (s < 8) ? (kb + s) : (16 + kb + (s - 8));
            b0[s] = bp[(size_t)k * KCODES];
            b1[s] = bp[(size_t)(32 + k) * KCODES];
        }

        v8f acc = {};
        acc = __builtin_amdgcn_wmma_f32_16x16x32_f16(false, a0, false, b0,
                                                     (short)0, acc, false, false);
        acc = __builtin_amdgcn_wmma_f32_16x16x32_f16(false, a1, false, b1,
                                                     (short)0, acc, false, false);

        float cn = lcn[tile * 16 + laneN];
        int code = tile * 16 + laneN;
        #pragma unroll
        for (int r = 0; r < 8; ++r) {
            float dist = rn[r] + cn - 2.0f * acc[r];   // = ||z - c||^2
            bool c = dist < minD[r];
            minD[r] = c ? dist : minD[r];
            minI[r] = c ? code : minI[r];
        }
    }

    // 16-lane min-reduce within each half-wave (xor<16 stays in half)
    #pragma unroll
    for (int off = 8; off >= 1; off >>= 1) {
        #pragma unroll
        for (int r = 0; r < 8; ++r) {
            float oD = __shfl_xor(minD[r], off, 32);
            int   oI = __shfl_xor(minI[r], off, 32);
            bool c = oD < minD[r];
            minD[r] = c ? oD : minD[r];
            minI[r] = c ? oI : minI[r];
        }
    }
    if (laneN == 0) {
        float s = 0.f;
        #pragma unroll
        for (int r = 0; r < 8; ++r) {
            int row = m0 + r + 8 * halfSel;
            idxOut[row] = minI[r];
            idxF[row] = (float)minI[r];
            s += minD[r];
        }
        atomicAdd(lossSum, s);
    }
}

// ---------------------------------------------------------------- gather quant -> q [B][D][T4] (f32)
__global__ __launch_bounds__(256)
void gather_kernel(const int* __restrict__ idx, const float* __restrict__ cb,
                   float* __restrict__ q) {
    int gid = blockIdx.x * 256 + threadIdx.x;    // (b*64 + d)*4096 + t
    int t = gid & (T4 - 1);
    int bd = gid >> 12;
    int d = bd & (ZDIM - 1);
    int b = bd >> 6;
    int ci = idx[b * T4 + t];
    q[gid] = cb[ci * ZDIM + d];
}

// ---------------------------------------------------------------- deconv1: 64->64, K4, s2, p1 + lrelu
// grid.x over (b, o32, t); grid.y = o-group (2)
__global__ __launch_bounds__(256)
void deconv1_kernel(const float* __restrict__ q, const float* __restrict__ w,
                    const float* __restrict__ bias, float* __restrict__ out) {
    __shared__ float lw[ZDIM * 32 * 4];          // 32 KB: [i][o32][k]
    int tid = threadIdx.x;
    int og = blockIdx.y;
    for (int id = tid; id < ZDIM * 32 * 4; id += 256) {
        int i = id >> 7, rem = id & 127;
        int o32 = rem >> 2, k = rem & 3;
        lw[id] = w[(i * 64 + og * 32 + o32) * 4 + k];
    }
    __syncthreads();
    int gid = blockIdx.x * 256 + tid;            // (b*32 + o32)*8192 + t
    int t = gid & (T2 - 1);
    int bo = gid >> 13;
    int o32 = bo & 31;
    int b = bo >> 5;
    int o = og * 32 + o32;
    const float* qb = q + b * ZDIM * T4;
    float acc = bias[o];
    if (t & 1) {                                 // taps k=0 (s=(t+1)/2), k=2 (s=(t-1)/2)
        int sA = (t + 1) >> 1, sB = (t - 1) >> 1;
        bool okA = sA < T4;
        #pragma unroll 4
        for (int i = 0; i < ZDIM; ++i) {
            const float* wp = &lw[i * 128 + o32 * 4];
            if (okA) acc = fmaf(wp[0], qb[i * T4 + sA], acc);
            acc = fmaf(wp[2], qb[i * T4 + sB], acc);
        }
    } else {                                     // taps k=1 (s=t/2), k=3 (s=t/2-1)
        int sA = t >> 1, sB = sA - 1;
        bool okB = sB >= 0;
        #pragma unroll 4
        for (int i = 0; i < ZDIM; ++i) {
            const float* wp = &lw[i * 128 + o32 * 4];
            acc = fmaf(wp[1], qb[i * T4 + sA], acc);
            if (okB) acc = fmaf(wp[3], qb[i * T4 + sB], acc);
        }
    }
    out[(b * ZDIM + o) * T2 + t] = lrelu(acc);
}

// ---------------------------------------------------------------- deconv2: 64->32, K4, s2, p1 + lrelu
__global__ __launch_bounds__(256)
void deconv2_kernel(const float* __restrict__ h3, const float* __restrict__ w,
                    const float* __restrict__ bias, float* __restrict__ out) {
    __shared__ float lw[ZDIM * HID * 4];         // 32 KB: [i][o][k] native layout
    int tid = threadIdx.x;
    for (int id = tid; id < ZDIM * HID * 4; id += 256) lw[id] = w[id];
    __syncthreads();
    int gid = blockIdx.x * 256 + tid;            // (b*32 + o)*16384 + t
    int t = gid & (T_IN - 1);
    int bo = gid >> 14;
    int o = bo & (HID - 1);
    int b = bo >> 5;
    const float* hb = h3 + b * ZDIM * T2;
    float acc = bias[o];
    if (t & 1) {
        int sA = (t + 1) >> 1, sB = (t - 1) >> 1;
        bool okA = sA < T2;
        #pragma unroll 4
        for (int i = 0; i < ZDIM; ++i) {
            const float* wp = &lw[(i * HID + o) * 4];
            if (okA) acc = fmaf(wp[0], hb[i * T2 + sA], acc);
            acc = fmaf(wp[2], hb[i * T2 + sB], acc);
        }
    } else {
        int sA = t >> 1, sB = sA - 1;
        bool okB = sB >= 0;
        #pragma unroll 4
        for (int i = 0; i < ZDIM; ++i) {
            const float* wp = &lw[(i * HID + o) * 4];
            acc = fmaf(wp[1], hb[i * T2 + sA], acc);
            if (okB) acc = fmaf(wp[3], hb[i * T2 + sB], acc);
        }
    }
    out[(b * HID + o) * T_IN + t] = lrelu(acc);
}

// ---------------------------------------------------------------- out conv: 32->1, K7, p3 + tanh
__global__ __launch_bounds__(256)
void outconv_kernel(const float* __restrict__ h4, const float* __restrict__ w,
                    const float* __restrict__ bias, float* __restrict__ recon) {
    __shared__ float lw[HID * 7];
    int tid = threadIdx.x;
    for (int i = tid; i < HID * 7; i += 256) lw[i] = w[i];
    __syncthreads();
    int gid = blockIdx.x * 256 + tid;            // b*16384 + t
    int t = gid & (T_IN - 1);
    int b = gid >> 14;
    const float* hb = h4 + b * HID * T_IN;
    float acc = bias[0];
    #pragma unroll 4
    for (int i = 0; i < HID; ++i) {
        #pragma unroll
        for (int k = 0; k < 7; ++k) {
            int p = t + k - 3;
            float v = (p >= 0 && p < T_IN) ? hb[i * T_IN + p] : 0.f;
            acc = fmaf(lw[i * 7 + k], v, acc);
        }
    }
    recon[gid] = tanhf(acc);
}

// ---------------------------------------------------------------- finalize vq_loss
__global__ void finalize_loss(const float* __restrict__ lossSum, float* __restrict__ out) {
    if (threadIdx.x == 0)
        out[0] = 1.25f * lossSum[0] / (float)(NROWS * ZDIM);
}

// ================================================================ host
extern "C" void kernel_launch(void* const* d_in, const int* in_sizes, int n_in,
                              void* d_out, int out_size, void* d_ws, size_t ws_size,
                              hipStream_t stream) {
    const float* x        = (const float*)d_in[0];
    const float* conv1_w  = (const float*)d_in[1];
    const float* conv1_b  = (const float*)d_in[2];
    const float* conv2_w  = (const float*)d_in[3];
    const float* conv2_b  = (const float*)d_in[4];
    const float* conv3_w  = (const float*)d_in[5];
    const float* conv3_b  = (const float*)d_in[6];
    const float* codebook = (const float*)d_in[7];
    const float* dec1_w   = (const float*)d_in[8];
    const float* dec1_b   = (const float*)d_in[9];
    const float* dec2_w   = (const float*)d_in[10];
    const float* dec2_b   = (const float*)d_in[11];
    const float* out_w    = (const float*)d_in[12];
    const float* out_b    = (const float*)d_in[13];

    char* ws = (char*)d_ws;
    float*  buf1     = (float*)(ws);                              // h1 (32MB) then h3 (64MB)
    float*  buf2     = (float*)(ws + (64ull << 20));              // h2 (32MB) then h4 (64MB)
    float*  q        = (float*)(ws + (128ull << 20));             // 32MB
    half_t* zh       = (half_t*)(ws + (160ull << 20));            // 16MB
    float*  rowNorm  = (float*)(ws + (176ull << 20));             // 512KB
    half_t* cbT      = (half_t*)(ws + (177ull << 20));            // 128KB
    float*  codeNorm = (float*)(ws + (177ull << 20) + (256u << 10));
    int*    idx      = (int*)(ws + (178ull << 20));               // 512KB
    float*  lossSum  = (float*)(ws + (179ull << 20));

    float* recon = (float*)d_out;                  // [524288]
    float* lossO = recon + (B_ * T_IN);            // [1]
    float* idxF  = lossO + 1;                      // [131072] as float

    hipMemsetAsync(lossSum, 0, sizeof(float), stream);

    prep_codebook<<<4, 256, 0, stream>>>(codebook, cbT, codeNorm);
    conv1_kernel<<<(B_ * HID * T2) / 256, 256, 0, stream>>>(x, conv1_w, conv1_b, buf1);
    conv2_kernel<<<dim3(T4 / 16, 4, B_), 256, 0, stream>>>(buf1, conv2_w, conv2_b, buf2);
    conv3_kernel<<<dim3(T4 / 128, B_), 128, 0, stream>>>(buf2, conv3_w, conv3_b, zh, rowNorm);
    vq_kernel<<<NROWS / 128, 256, 0, stream>>>(zh, rowNorm, cbT, codeNorm, idx, idxF, lossSum);
    gather_kernel<<<(B_ * ZDIM * T4) / 256, 256, 0, stream>>>(idx, codebook, q);
    deconv1_kernel<<<dim3((B_ * 32 * T2) / 256, 2), 256, 0, stream>>>(q, dec1_w, dec1_b, buf1);
    deconv2_kernel<<<(B_ * HID * T_IN) / 256, 256, 0, stream>>>(buf1, dec2_w, dec2_b, buf2);
    outconv_kernel<<<(B_ * T_IN) / 256, 256, 0, stream>>>(buf2, out_w, out_b, recon);
    finalize_loss<<<1, 32, 0, stream>>>(lossSum, lossO);
}
